// MultiScaleRetention_30245159698610
// MI455X (gfx1250) — matrix operand order repaired and checked by reading, not compiled, not run on hardware
//
#include <hip/hip_runtime.h>
#include <hip/hip_bf16.h>
#include <stdint.h>

// ---------------------------------------------------------------------------
// Types for CDNA5 WMMA (gfx1250, wave32)
// ---------------------------------------------------------------------------
typedef __bf16 bf16;
typedef __attribute__((ext_vector_type(16))) __bf16 v16bf;
typedef __attribute__((ext_vector_type(8)))  __bf16 v8bf;
typedef __attribute__((ext_vector_type(8)))  float  v8f;

union FragBF { v16bf v; v8bf h[2]; };

__device__ __forceinline__ v8bf zero8() {
  v8bf z;
#pragma unroll
  for (int i = 0; i < 8; ++i) z[i] = (bf16)0.f;
  return z;
}

__device__ __forceinline__ v8f wmma_bf(const FragBF& a, const FragBF& b, v8f c) {
  // D(f32 16x16) = A(bf16 16x32) * B(bf16 32x16) + C
  return __builtin_amdgcn_wmma_f32_16x16x32_bf16(false, a.v, false, b.v,
                                                 (short)0, c, false, false);
}

__device__ __forceinline__ void lds_fence_wave() {
  // intra-wave LDS RAW visibility: DS pipe is in-order; wait outstanding DS ops
  __builtin_amdgcn_wave_barrier();
  asm volatile("s_wait_dscnt 0" ::: "memory");
  __builtin_amdgcn_wave_barrier();
}

#define ROWS (256LL*1024LL)          // B*T = 262144

// ---------------------------------------------------------------------------
// Workspace layout (bytes). All sizes 256B aligned by construction.
// ---------------------------------------------------------------------------
static const size_t OFF_HBF      = 0;                                  // bf16 262144x256
static const size_t OFF_QKV      = OFF_HBF      + 262144ULL*256*2;     // bf16 262144x768
static const size_t OFF_QB       = OFF_QKV      + 262144ULL*768*2;     // bf16 [B][H][T][16]
static const size_t OFF_KB       = OFF_QB       + 262144ULL*256*2;
static const size_t OFF_VB       = OFF_KB       + 262144ULL*256*2;
static const size_t OFF_GATED    = OFF_VB       + 262144ULL*256*2;     // bf16 silu(H Wg^T)
static const size_t OFF_RETA     = OFF_GATED    + 262144ULL*256*2;     // f32 [B][T][H][16]
static const size_t OFF_RETB     = OFF_RETA     + 262144ULL*256*4;     // bf16 [B][H][T][16]
static const size_t OFF_XG       = OFF_RETB     + 262144ULL*256*2;     // bf16 gated*normed
static const size_t OFF_WQKV     = OFF_XG       + 262144ULL*256*2;     // bf16 weights:
static const size_t OFF_WGATED   = OFF_WQKV     + 768ULL*256*2;
static const size_t OFF_WPROJ    = OFF_WGATED   + 256ULL*256*2;
static const size_t OFF_WGATPROJ = OFF_WPROJ    + 256ULL*256*2;
static const size_t OFF_WGATLIN  = OFF_WGATPROJ + 256ULL*1024*2;
static const size_t OFF_WGATT    = OFF_WGATLIN  + 512ULL*4096*2;       // [h][o=256][f=4096]
static const size_t OFF_WIH      = OFF_WGATT    + 16ULL*256*4096*2;
static const size_t OFF_WHH      = OFF_WIH      + 1536ULL*512*2;
static const size_t OFF_GATPROJ  = OFF_WHH      + 1536ULL*512*2;       // bf16 4096x256
static const size_t OFF_WH       = OFF_GATPROJ  + 4096ULL*256*2;       // f32 16x256x256
static const size_t OFF_LB       = OFF_WH       + 16ULL*256*256*4;     // f32 16x256
static const size_t OFF_RB       = OFF_LB       + 16ULL*256*4;
static const size_t OFF_GATOUT   = OFF_RB       + 16ULL*256*4;         // bf16 256x4096
static const size_t OFF_TEMP     = OFF_GATOUT   + 256ULL*4096*2;       // f32 256x512
static const size_t OFF_TEMPBF   = OFF_TEMP     + 256ULL*512*4;
static const size_t OFF_GI       = OFF_TEMPBF   + 256ULL*512*2;        // f32 256x1536
static const size_t OFF_GH       = OFF_GI       + 256ULL*1536*4;
static const size_t OFF_STATE    = OFF_GH       + 256ULL*1536*4;       // f32 256x512
static const size_t OFF_STATEBF  = OFF_STATE    + 256ULL*512*4;
static const size_t OFF_XTAB     = OFF_STATEBF  + 256ULL*512*2;        // f32x4 1024x128

// ---------------------------------------------------------------------------
// Generic bf16 WMMA GEMM:  D[M x N] = A[M x K] * W[N x K]^T  (+bias, silu)
// Block = 256 thr (8 waves), block tile = 128(M) x 64(N), wave tile = 16x64.
// M % 128 == 0, N % 64 == 0, K % 32 == 0 for all call sites.
// FLAGS: 1 = SiLU epilogue, 2 = bf16 output (else f32), 4 = add f32 bias[col]
// ---------------------------------------------------------------------------
template <int FLAGS>
__global__ __launch_bounds__(256) void gemm_wmma(
    const bf16* __restrict__ A, int lda, long long strideA,
    const bf16* __restrict__ W, int ldw, long long strideW,
    void* __restrict__ Dv, int ldd, long long strideD,
    int K, const float* __restrict__ bias)
{
  const int lane = threadIdx.x & 31;
  const int wave = threadIdx.x >> 5;
  const int bz = blockIdx.z;
  const bf16* Ab = A + strideA * bz;
  const bf16* Wb = W + strideW * bz;

  const int m0 = blockIdx.x * 128 + wave * 16;
  const int n0 = blockIdx.y * 64;
  const int lm  = lane & 15;
  const int kb  = (lane < 16) ? 0 : 8;    // unused; A-frag also loads at klo
  const int klo = (lane < 16) ? 0 : 16;   // B-frag K base within 32
  const long long rowA = (long long)(m0 + lm) * lda;

  v8f acc[4];
  const v8f zacc = {0.f,0.f,0.f,0.f,0.f,0.f,0.f,0.f};
#pragma unroll
  for (int nt = 0; nt < 4; ++nt) acc[nt] = zacc;

  for (int k0 = 0; k0 < K; k0 += 32) {
    if (k0 + 32 < K)                       // emits global_prefetch_b8
      __builtin_prefetch(Ab + rowA + k0 + 32, 0, 0);
    FragBF a;
    a.v = *(const v16bf*)(Ab + rowA + k0 + klo);
#pragma unroll
    for (int nt = 0; nt < 4; ++nt) {
      const int col = n0 + nt * 16 + lm;
      FragBF b;
      b.v = *(const v16bf*)(Wb + (long long)col * ldw + k0 + klo);
      acc[nt] = wmma_bf(a, b, acc[nt]);
    }
  }

#pragma unroll
  for (int nt = 0; nt < 4; ++nt) {
    const int col = n0 + nt * 16 + lm;
    float bv = 0.f;
    if (FLAGS & 4) bv = bias[col];
#pragma unroll
    for (int r = 0; r < 8; ++r) {
      const int mrow = m0 + ((lane < 16) ? r : 8 + r);
      float v = acc[nt][r] + bv;
      if (FLAGS & 1) v = v / (1.f + __expf(-v));   // SiLU
      const long long di = strideD * bz + (long long)mrow * ldd + col;
      if (FLAGS & 2) ((bf16*)Dv)[di] = (bf16)v;
      else           ((float*)Dv)[di] = v;
    }
  }
}

// ---------------------------------------------------------------------------
// Elementwise / conversion kernels
// ---------------------------------------------------------------------------
__global__ void cvt_f32_bf16_k(const float* __restrict__ s, bf16* __restrict__ d,
                               long long n) {
  long long i = (long long)blockIdx.x * 256 + threadIdx.x;
  if (i < n) d[i] = (bf16)s[i];
}

// W_gat [h][f=4096][o=256] f32  ->  [h][o=256][f=4096] bf16
__global__ void tr_wgat_k(const float* __restrict__ s, bf16* __restrict__ d) {
  long long i = (long long)blockIdx.x * 256 + threadIdx.x;
  if (i >= 16LL * 4096 * 256) return;
  int o = (int)(i & 255);
  long long rem = i >> 8;
  int f = (int)(rem & 4095);
  int hh = (int)(rem >> 12);
  d[((long long)hh * 256 + o) * 4096 + f] = (bf16)s[i];
}

__global__ void init_state_k(float* st, bf16* stbf) {
  int i = blockIdx.x * 256 + threadIdx.x;
  if (i < 256 * 512) { st[i] = 0.f; stbf[i] = (bf16)0.f; }
}

__global__ void copy_state_k(const float* st, float* dst) {
  int i = blockIdx.x * 256 + threadIdx.x;
  if (i < 256 * 512) dst[i] = st[i];
}

// xpos trig/scale table over (t,j): {scale, 1/scale, sin, cos}
__global__ void xpos_tab_k(float4* __restrict__ tab) {
  int i = blockIdx.x * 256 + threadIdx.x;    // t*128 + j
  if (i >= 1024 * 128) return;
  const int t = i >> 7, j = i & 127;
  const float bj = (2.f * j + 102.4f) / 358.4f;          // (2j + 0.4d)/(1.4d)
  const float sc = __powf(bj, (float)t / 512.f);
  const float invf = __powf(10000.f, -(float)j / 128.f);
  const float ang = (float)t * invf;
  float4 r; r.x = sc; r.y = 1.f / sc; r.z = __sinf(ang); r.w = __cosf(ang);
  tab[i] = r;
}

// xpos rotary + head split: qkv bf16 [B*T][768] -> q/k/v bf16 [B][H][T][16]
__global__ void xpos_split_k(const bf16* __restrict__ qkv,
                             const float4* __restrict__ tab,
                             bf16* __restrict__ qo, bf16* __restrict__ ko,
                             bf16* __restrict__ vo) {
  long long idx = (long long)blockIdx.x * 256 + threadIdx.x;
  if (idx >= ROWS * 128) return;
  const int j = (int)(idx & 127);
  const long long row = idx >> 7;            // b*T + t
  const int t = (int)(row & 1023);
  const int b = (int)(row >> 10);
  const bf16* base = qkv + row * 768;
  const float q1 = (float)base[2*j],     q2 = (float)base[2*j+1];
  const float k1 = (float)base[256+2*j], k2 = (float)base[256+2*j+1];
  const float v1 = (float)base[512+2*j], v2 = (float)base[512+2*j+1];
  const float4 tb = tab[t * 128 + j];
  const float qs = tb.z * tb.x, qc = tb.w * tb.x;        // q: downscale=False
  const float ks = tb.z * tb.y, kc = tb.w * tb.y;        // k: downscale=True
  const float qa = q1 * qc - q2 * qs;
  const float qb = q2 * qc + q1 * qs;
  const float ka = k1 * kc - k2 * ks;
  const float kbv = k2 * kc + k1 * ks;
  const int f0 = 2 * j, f1 = 2 * j + 1;
  const int h0 = f0 >> 4, d0 = f0 & 15;
  const int h1 = f1 >> 4, d1 = f1 & 15;
  const long long o0 = (((long long)(b * 16 + h0)) * 1024 + t) * 16 + d0;
  const long long o1 = (((long long)(b * 16 + h1)) * 1024 + t) * 16 + d1;
  qo[o0] = (bf16)qa;  qo[o1] = (bf16)qb;
  ko[o0] = (bf16)ka;  ko[o1] = (bf16)kbv;
  vo[o0] = (bf16)v1;  vo[o1] = (bf16)v2;
}

// ---------------------------------------------------------------------------
// Retention scan. 1 wave per (b,h); loops all 16 chunks with the 16x16 KV
// state held in registers. All decay powers are separable products of a few
// per-lane constants -> only 3 __powf per wave total; hot loop is pure FMA +
// WMMA. Per-wave private LDS slice: rS(64x64 bf16), kT/vT(16x64 bf16),
// tmp(16x16 f32).
// ---------------------------------------------------------------------------
__global__ __launch_bounds__(128) void retention_kernel(
    const bf16* __restrict__ q, const bf16* __restrict__ k,
    const bf16* __restrict__ v,
    float* __restrict__ retA,     // [B][T][H][16] f32
    bf16*  __restrict__ retB,     // [B][H][T][16] bf16
    float* __restrict__ kv_out)   // [B][H][16(d)][16(v)] f32
{
  __shared__ char lds_raw[4 * 13312];
  const int lane = threadIdx.x & 31;
  const int wave = threadIdx.x >> 5;
  const int wid = blockIdx.x * 4 + wave;       // 0..4095
  const int b = wid >> 4;
  const int h = wid & 15;
  char* base = lds_raw + wave * 13312;
  bf16*  rS  = (bf16*)base;                    // 64*64
  bf16*  kT  = (bf16*)(base + 8192);           // 16*64
  bf16*  vT  = (bf16*)(base + 10240);          // 16*64
  float* tmp = (float*)(base + 12288);         // 16*16

  const float gamma = 1.f - exp2f(-5.f - (float)h);
  const float scale = 0.25f;                   // qk_h^-0.5
  const int kb  = (lane < 16) ? 0 : 8;
  const int klo = (lane < 16) ? 0 : 16;
  const int lm = lane & 15;
  const int rrbase = (lane < 16) ? 0 : 8;
  const v8f zacc = {0.f,0.f,0.f,0.f,0.f,0.f,0.f,0.f};

  // ---- hoisted decay-power constants (all gamma^e are separable) ----
  const float ginv = 1.f / gamma;
  float grr[8], girr[8];
  grr[0] = 1.f; girr[0] = 1.f;
#pragma unroll
  for (int i = 1; i < 8; ++i) { grr[i] = grr[i-1] * gamma; girr[i] = girr[i-1] * ginv; }
  const float g8   = grr[7] * gamma;           // gamma^8
  const float g16f = g8 * g8;                  // gamma^16
  const float gi16 = 1.f / g16f;               // gamma^-16
  float g16p[7];                               // gamma^(16*(k-3)), k=0..6
  g16p[3] = 1.f;
  g16p[4] = g16f;  g16p[5] = g16p[4] * g16f;  g16p[6] = g16p[5] * g16f;
  g16p[2] = gi16;  g16p[1] = g16p[2] * gi16;  g16p[0] = g16p[1] * gi16;
  const float cd    = g16p[6] * g16f;          // gamma^64 (chunk decay)
  const float pbase = __powf(gamma, (float)(rrbase - lm)); // gamma^(rrbase-lm)
  const float gS    = (lane < 16) ? 1.f : g8;  // gamma^rrbase
  const float gin0  = __powf(gamma, (float)(63 - kb));     // k0 = 0
  const float gin32 = __powf(gamma, (float)(31 - kb));     // k0 = 32

  const long long hoff = ((long long)(b * 16 + h)) * (1024LL * 16);
  const bf16* qh = q + hoff;
  const bf16* kh = k + hoff;
  const bf16* vh = v + hoff;

  float s[16];                                 // state col v=lane (lanes<16), K=d
#pragma unroll
  for (int i = 0; i < 16; ++i) s[i] = 0.f;

  for (int ci = 0; ci < 16; ++ci) {
    // -- q A-fragments (K=16 padded to 32)
    FragBF aq[4];
#pragma unroll
    for (int mt = 0; mt < 4; ++mt) {
      const int c = ci * 64 + mt * 16 + lm;
      if (lane < 16) aq[mt].v = *(const v16bf*)(qh + (long long)c * 16);
      else { aq[mt].h[0] = zero8(); aq[mt].h[1] = zero8(); }
    }
    // -- ret = q k^T : B-fragment column c' == row c' of k (no transpose)
    v8f r[4][4];
#pragma unroll
    for (int nt = 0; nt < 4; ++nt) {
      FragBF bk;
      if (lane < 16) bk.v = *(const v16bf*)(kh + (long long)(ci * 64 + nt * 16 + lm) * 16);
      else { bk.h[0] = zero8(); bk.h[1] = zero8(); }
#pragma unroll
      for (int mt = 0; mt < 4; ++mt) r[mt][nt] = wmma_bf(aq[mt], bk, zacc);
    }
    // -- apply scale * causal decay mask (gamma^(c-cc)), write bf16 to LDS
#pragma unroll
    for (int mt = 0; mt < 4; ++mt)
#pragma unroll
      for (int nt = 0; nt < 4; ++nt) {
        const float gmn = g16p[mt - nt + 3] * pbase * scale;
#pragma unroll
        for (int rr = 0; rr < 8; ++rr) {
          const int c  = mt * 16 + rrbase + rr;
          const int cc = nt * 16 + lm;
          const float val = (c >= cc) ? r[mt][nt][rr] * gmn * grr[rr] : 0.f;
          rS[c * 64 + cc] = (bf16)val;
        }
      }
    // -- stage k^T and v^T in LDS
    for (int idx = lane; idx < 1024; idx += 32) {
      const int c = idx >> 4, d = idx & 15;
      kT[d * 64 + c] = kh[(long long)(ci * 64 + c) * 16 + d];
      vT[d * 64 + c] = vh[(long long)(ci * 64 + c) * 16 + d];
    }
    lds_fence_wave();

    // -- cross = (q @ S) * inner_decay(gamma^(c+1)) * scale
    FragBF bs;
    if (lane < 16) {
#pragma unroll
      for (int i = 0; i < 8; ++i) { bs.h[0][i] = (bf16)s[i]; bs.h[1][i] = (bf16)s[8 + i]; }
    } else { bs.h[0] = zero8(); bs.h[1] = zero8(); }
    v8f o[4];
#pragma unroll
    for (int mt = 0; mt < 4; ++mt) {
      o[mt] = wmma_bf(aq[mt], bs, zacc);
      const float gcr = scale * gamma * g16p[mt + 3] * gS;
#pragma unroll
      for (int rr = 0; rr < 8; ++rr) o[mt][rr] *= gcr * grr[rr];
    }
    // -- inner = masked_scores @ v  (accumulate into o)
#pragma unroll
    for (int mt = 0; mt < 4; ++mt)
#pragma unroll
      for (int kk = 0; kk < 2; ++kk) {
        const int k0 = kk * 32;
        FragBF a, bv;
        const bf16* rrow = rS + (mt * 16 + lm) * 64 + k0;
        a.v = *(const v16bf*)(rrow + klo);
        bv.v = *(const v16bf*)(vT + lm * 64 + k0 + klo);
        o[mt] = wmma_bf(a, bv, o[mt]);
      }
    // -- store chunk output in both layouts
#pragma unroll
    for (int mt = 0; mt < 4; ++mt)
#pragma unroll
      for (int rr = 0; rr < 8; ++rr) {
        const int c = mt * 16 + rrbase + rr;
        const int t = ci * 64 + c;
        const float val = o[mt][rr];
        retA[(((long long)b * 1024 + t) * 16 + h) * 16 + lm] = val;
        retB[hoff + (long long)t * 16 + lm] = (bf16)val;
      }
    // -- state update: U[v][d] = sum_c v[c][v] * gamma^(63-c) * k[c][d]
    v8f u = zacc;
#pragma unroll
    for (int kk = 0; kk < 2; ++kk) {
      const int k0 = kk * 32;
      const float gb = (kk == 0) ? gin0 : gin32;   // gamma^(63-k0-kb)
      FragBF a, bkt;
      const bf16* vr = vT + lm * 64;
      v8bf x0 = *(const v8bf*)(vr + k0 + kb);
      v8bf x1 = *(const v8bf*)(vr + k0 + kb + 16);
#pragma unroll
      for (int e = 0; e < 8; ++e) {
        a.h[0][e] = (bf16)((float)x0[e] * gb * girr[e]);          // 63-(k0+kb+e)
        a.h[1][e] = (bf16)((float)x1[e] * gb * gi16 * girr[e]);   // 63-(k0+kb+16+e)
      }
      bkt.h[0] = *(const v8bf*)(kT + lm * 64 + k0 + kb);
      bkt.h[1] = *(const v8bf*)(kT + lm * 64 + k0 + kb + 16);
      u = wmma_bf(a, bkt, u);
    }
    // transpose U (lane=d, vgpr=v) -> state layout (lane=v, idx=d) via LDS
#pragma unroll
    for (int rr = 0; rr < 8; ++rr) {
      const int vv = (lane < 16) ? rr : 8 + rr;
      tmp[vv * 16 + lm] = u[rr];
    }
    lds_fence_wave();
    if (lane < 16) {
#pragma unroll
      for (int d = 0; d < 16; ++d) s[d] = cd * s[d] + tmp[lane * 16 + d];
    }
    lds_fence_wave();
  }
  if (lane < 16) {
#pragma unroll
    for (int d = 0; d < 16; ++d)
      kv_out[(((long long)(b * 16 + h)) * 16 + d) * 16 + lane] = s[d];
  }
}

// ---------------------------------------------------------------------------
// GAT: per-row dots with a_gat halves (wave per (h2,row))
// ---------------------------------------------------------------------------
__global__ __launch_bounds__(128) void gat_dot_k(
    const float* __restrict__ Wh, const float* __restrict__ a_gat,
    float* __restrict__ L, float* __restrict__ R) {
  const int lane = threadIdx.x & 31;
  const int wave = threadIdx.x >> 5;
  const int id = blockIdx.x * 4 + wave;       // 0..4095
  const int h2 = id >> 8, row = id & 255;
  const float* wr = Wh + ((long long)h2 * 256 + row) * 256;
  const float* ag = a_gat + h2 * 512;
  float sl = 0.f, sr = 0.f;
  for (int o = lane; o < 256; o += 32) {
    const float w = wr[o];
    sl += w * ag[o];
    sr += w * ag[256 + o];
  }
#pragma unroll
  for (int off = 16; off > 0; off >>= 1) {
    sl += __shfl_xor(sl, off, 32);
    sr += __shfl_xor(sr, off, 32);
  }
  if (lane == 0) { L[h2 * 256 + row] = sl; R[h2 * 256 + row] = sr; }
}

// softmax(32) + att@Wh + elu + relu, one block per (graph, gat-head)
__global__ __launch_bounds__(256) void gat_att_k(
    const float* __restrict__ Wh, const float* __restrict__ L,
    const float* __restrict__ R, bf16* __restrict__ gout) {
  __shared__ float att[1024];
  __shared__ float lb[32], rb[32];
  const int g = blockIdx.x, h2 = blockIdx.y;
  const int tid = threadIdx.x;
  if (tid < 32) {
    lb[tid] = L[h2 * 256 + g * 32 + tid];
    rb[tid] = R[h2 * 256 + g * 32 + tid];
  }
  __syncthreads();
  if (tid < 32) {
    float mx = -3.4e38f;
#pragma unroll
    for (int m = 0; m < 32; ++m) mx = fmaxf(mx, lb[tid] + rb[m]);
    float sm = 0.f;
#pragma unroll
    for (int m = 0; m < 32; ++m) {
      const float e = __expf(lb[tid] + rb[m] - mx);
      att[tid * 32 + m] = e; sm += e;
    }
    const float inv = 1.f / sm;
#pragma unroll
    for (int m = 0; m < 32; ++m) att[tid * 32 + m] *= inv;
  }
  __syncthreads();
  const int o = tid;
  float acc[32];
#pragma unroll
  for (int n = 0; n < 32; ++n) acc[n] = 0.f;
  const float* wb = Wh + (long long)h2 * 65536 + (long long)g * 8192;
  for (int m = 0; m < 32; ++m) {
    const float w = wb[m * 256 + o];
#pragma unroll
    for (int n = 0; n < 32; ++n) acc[n] += att[n * 32 + m] * w;
  }
  for (int n = 0; n < 32; ++n) {
    const float vv = acc[n];
    const float e = vv > 0.f ? vv : (__expf(vv) - 1.f);   // elu
    const float rl = e > 0.f ? e : 0.f;                   // relu
    gout[((long long)(g * 32 + n)) * 4096 + h2 * 256 + o] = (bf16)rl;
  }
}

// GRU gate combine; ci==0 takes temp directly (per reference jnp.where)
__global__ void gru_k(const float* __restrict__ gi, const float* __restrict__ gh,
                      const float* __restrict__ temp, float* __restrict__ st,
                      bf16* __restrict__ stbf, int ci) {
  const int i = blockIdx.x * 256 + threadIdx.x;
  if (i >= 256 * 512) return;
  const int row = i >> 9, col = i & 511;
  const long long base = (long long)row * 1536;
  const float ir = gi[base + col],      hr = gh[base + col];
  const float iz = gi[base + 512 + col], hz = gh[base + 512 + col];
  const float in_ = gi[base + 1024 + col], hn = gh[base + 1024 + col];
  const float h0 = st[i];
  const float r = 1.f / (1.f + __expf(-(ir + hr)));
  const float z = 1.f / (1.f + __expf(-(iz + hz)));
  const float n = tanhf(in_ + r * hn);
  float hnew = (1.f - z) * n + z * h0;
  if (ci == 0) hnew = temp[i];
  st[i] = hnew;
  stbf[i] = (bf16)hnew;
}

// per-(b,t,h) group norm over v_h=16 then multiply by silu-gated (already bf16)
__global__ void gnorm_gate_k(const float* __restrict__ retA,
                             const bf16* __restrict__ gated,
                             bf16* __restrict__ Xg) {
  const long long gid = (long long)blockIdx.x * 256 + threadIdx.x;
  if (gid >= ROWS * 16) return;
  const long long base = gid * 16;
  float x[16]; float mu = 0.f;
#pragma unroll
  for (int j = 0; j < 16; ++j) { x[j] = retA[base + j]; mu += x[j]; }
  mu *= (1.f / 16.f);
  float var = 0.f;
#pragma unroll
  for (int j = 0; j < 16; ++j) { const float d = x[j] - mu; var += d * d; }
  var *= (1.f / 16.f);
  const float inv = rsqrtf(var + 1e-5f);
#pragma unroll
  for (int j = 0; j < 16; ++j)
    Xg[base + j] = (bf16)((float)gated[base + j] * ((x[j] - mu) * inv));
}

// ---------------------------------------------------------------------------
// Host orchestration
// ---------------------------------------------------------------------------
extern "C" void kernel_launch(void* const* d_in, const int* in_sizes, int n_in,
                              void* d_out, int out_size, void* d_ws, size_t ws_size,
                              hipStream_t stream) {
  const float* hidden   = (const float*)d_in[0];
  const float* W_qkv    = (const float*)d_in[1];
  const float* W_gated  = (const float*)d_in[2];
  const float* W_proj   = (const float*)d_in[3];
  const float* W_gatprj = (const float*)d_in[4];
  const float* W_gatlin = (const float*)d_in[5];
  const float* W_gat    = (const float*)d_in[6];
  const float* a_gat    = (const float*)d_in[7];
  const float* w_ih     = (const float*)d_in[8];
  const float* w_hh     = (const float*)d_in[9];
  const float* b_ih     = (const float*)d_in[10];
  const float* b_hh     = (const float*)d_in[11];
  (void)in_sizes; (void)n_in; (void)out_size; (void)ws_size;

  char* ws = (char*)d_ws;
  bf16*  Hbf   = (bf16*)(ws + OFF_HBF);
  bf16*  qkvbf = (bf16*)(ws + OFF_QKV);
  bf16*  qb    = (bf16*)(ws + OFF_QB);
  bf16*  kbuf  = (bf16*)(ws + OFF_KB);
  bf16*  vb    = (bf16*)(ws + OFF_VB);
  bf16*  gated = (bf16*)(ws + OFF_GATED);
  float* retA  = (float*)(ws + OFF_RETA);
  bf16*  retB  = (bf16*)(ws + OFF_RETB);
  bf16*  Xg    = (bf16*)(ws + OFF_XG);
  bf16*  Wqkvb = (bf16*)(ws + OFF_WQKV);
  bf16*  Wgtdb = (bf16*)(ws + OFF_WGATED);
  bf16*  Wprjb = (bf16*)(ws + OFF_WPROJ);
  bf16*  Wgpb  = (bf16*)(ws + OFF_WGATPROJ);
  bf16*  Wglb  = (bf16*)(ws + OFF_WGATLIN);
  bf16*  WgTb  = (bf16*)(ws + OFF_WGATT);
  bf16*  wihb  = (bf16*)(ws + OFF_WIH);
  bf16*  whhb  = (bf16*)(ws + OFF_WHH);
  bf16*  gatprj= (bf16*)(ws + OFF_GATPROJ);
  float* Wh    = (float*)(ws + OFF_WH);
  float* Lb    = (float*)(ws + OFF_LB);
  float* Rb    = (float*)(ws + OFF_RB);
  bf16*  gatout= (bf16*)(ws + OFF_GATOUT);
  float* temp  = (float*)(ws + OFF_TEMP);
  bf16*  tempbf= (bf16*)(ws + OFF_TEMPBF);
  float* gi    = (float*)(ws + OFF_GI);
  float* gh    = (float*)(ws + OFF_GH);
  float* state = (float*)(ws + OFF_STATE);
  bf16*  statebf=(bf16*)(ws + OFF_STATEBF);
  float4* xtab = (float4*)(ws + OFF_XTAB);

  float* out0     = (float*)d_out;               // (B,T,256)
  float* kvout    = out0 + 67108864LL;           // (B,H,16,16)
  float* gatfinal = out0 + 68157440LL;           // (8,32,512)

#define CVT(src, dst, n) \
  cvt_f32_bf16_k<<<dim3((unsigned)(((n) + 255) / 256)), dim3(256), 0, stream>>>(src, dst, (long long)(n))

  // 1) precision conversion of activations & weights; xpos tables
  CVT(hidden, Hbf, 67108864LL);
  CVT(W_qkv,  Wqkvb, 196608LL);
  CVT(W_gated, Wgtdb, 65536LL);
  CVT(W_proj,  Wprjb, 65536LL);
  CVT(W_gatprj, Wgpb, 262144LL);
  CVT(W_gatlin, Wglb, 2097152LL);
  CVT(w_ih, wihb, 786432LL);
  CVT(w_hh, whhb, 786432LL);
  tr_wgat_k<<<dim3(65536), dim3(256), 0, stream>>>(W_gat, WgTb);
  xpos_tab_k<<<dim3(512), dim3(256), 0, stream>>>(xtab);

  // 2) qkv projection and silu gate (WMMA bf16)
  gemm_wmma<2><<<dim3(2048, 12, 1), 256, 0, stream>>>(Hbf, 256, 0, Wqkvb, 256, 0,
                                                      qkvbf, 768, 0, 256, nullptr);
  gemm_wmma<3><<<dim3(2048, 4, 1), 256, 0, stream>>>(Hbf, 256, 0, Wgtdb, 256, 0,
                                                     gated, 256, 0, 256, nullptr);

  // 3) xpos rotary + head split
  xpos_split_k<<<dim3(131072), dim3(256), 0, stream>>>(qkvbf, xtab, qb, kbuf, vb);

  // 4) full retention scan (state in registers, WMMA inner kernels)
  retention_kernel<<<dim3(1024), dim3(128), 0, stream>>>(qb, kbuf, vb, retA, retB, kvout);

  // 5) GAT + GRU chain, sequential over chunks
  init_state_k<<<dim3(512), dim3(256), 0, stream>>>(state, statebf);
  for (int ci = 0; ci < 16; ++ci) {
    // gat_in = out_chunk(4096x1024) @ W_gat_proj^T -> (4096x256) == (256x4096)
    gemm_wmma<2><<<dim3(32, 4, 1), 256, 0, stream>>>(
        retB + (long long)ci * 1024, 16384, 0, Wgpb, 1024, 0,
        gatprj, 256, 0, 1024, nullptr);
    // Wh[h2] = x(256x4096) @ W_gat[h2]  (batched over 16 heads via grid.z)
    gemm_wmma<0><<<dim3(2, 4, 16), 256, 0, stream>>>(
        gatprj, 4096, 0, WgTb, 4096, 1048576LL, Wh, 256, 65536LL, 4096, nullptr);
    gat_dot_k<<<dim3(1024), dim3(128), 0, stream>>>(Wh, a_gat, Lb, Rb);
    gat_att_k<<<dim3(8, 16, 1), 256, 0, stream>>>(Wh, Lb, Rb, gatout);
    // temp = gat_out(256x4096) @ W_gat_linear^T -> (256x512)
    gemm_wmma<0><<<dim3(2, 8, 1), 256, 0, stream>>>(
        gatout, 4096, 0, Wglb, 4096, 0, temp, 512, 0, 4096, nullptr);
    cvt_f32_bf16_k<<<dim3(512), dim3(256), 0, stream>>>(temp, tempbf, 131072LL);
    // GRU gates
    gemm_wmma<4><<<dim3(2, 24, 1), 256, 0, stream>>>(
        tempbf, 512, 0, wihb, 512, 0, gi, 1536, 0, 512, b_ih);
    gemm_wmma<4><<<dim3(2, 24, 1), 256, 0, stream>>>(
        statebf, 512, 0, whhb, 512, 0, gh, 1536, 0, 512, b_hh);
    gru_k<<<dim3(512), dim3(256), 0, stream>>>(gi, gh, temp, state, statebf, ci);
  }
  copy_state_k<<<dim3(512), dim3(256), 0, stream>>>(state, gatfinal);

  // 6) group-norm over heads, gate, and final projection
  gnorm_gate_k<<<dim3(16384), dim3(256), 0, stream>>>(retA, gated, Xg);
  gemm_wmma<0><<<dim3(2048, 4, 1), 256, 0, stream>>>(
      Xg, 256, 0, Wprjb, 256, 0, out0, 256, 0, 256, nullptr);
#undef CVT
}
